// NeuronNemotronMamba2Layer_25890062860802
// MI455X (gfx1250) — matrix operand order repaired and checked by reading, not compiled, not run on hardware
//
#include <hip/hip_runtime.h>
#include <hip/hip_bf16.h>
#include <math.h>

// ---- problem constants (match reference) ----
#define Hh 64
#define Pp 64
#define Nn 128
#define Gg 8
#define Kk 4
#define DMODEL 4096
#define Ii (Hh * Pp)                 // 4096
#define GNc (Gg * Nn)                // 1024
#define CONVD (Ii + 2 * GNc)         // 6144
#define PROJD (Ii + CONVD + Hh)      // 10304
#define Bb 2
#define Ll 2048
#define EPSI 1e-5f
#define GRPSZ (Ii / Gg)              // 512

typedef __attribute__((ext_vector_type(16))) __bf16 v16bf;
typedef __attribute__((ext_vector_type(8)))  float  v8f;
typedef __attribute__((ext_vector_type(4)))  int    v4i;

union Frag16 { v16bf v; v4i q[2]; };

// ---- TDM descriptor vector types (probe-confirmed builtin signature) ----
typedef __attribute__((ext_vector_type(4))) unsigned int tdm_g0_t;
typedef __attribute__((ext_vector_type(8))) int          tdm_g1_t;
typedef __attribute__((ext_vector_type(4))) int          tdm_g2_t;

#if defined(__has_include)
#if __has_include(<hip/amd_detail/amd_gfx1250_TDM.h>)
#define TDM_6ARG 1
#endif
#endif
#ifndef TDM_6ARG
#define TDM_6ARG 0
#endif

// Issue one TDM 2D tile load: tilerows x 64 bf16 elements, global row stride = kdim
// elements, into LDS at ldsoff with 16B padding per 128B row (LDS row stride 144B).
__device__ __forceinline__ void tdm_issue_tile(const __bf16* gptr, int kdim, int totrows,
                                               int tilerows, unsigned ldsoff)
{
    unsigned long long ga = (unsigned long long)(const void*)gptr;
    tdm_g0_t g0;
    g0.x = 1u;                                  // count=1, is_restore=0, gather off
    g0.y = ldsoff;                              // lds_addr (bytes)
    g0.z = (unsigned)ga;                        // global_addr[31:0]
    g0.w = (unsigned)((ga >> 32) & 0x01FFFFFFull) | (2u << 30);  // addr[56:32], type=2
    tdm_g1_t g1;
    // word0: data_size=1(2B) @16, pad_enable @20, pad_interval=4(128B) @22, pad_amount=3(4 DW) @25
    g1[0] = (int)0x07110000u;
    g1[1] = (int)(((unsigned)kdim & 0xFFFFu) << 16);              // tensor_dim0[15:0] @48
    g1[2] = (int)(((unsigned)kdim >> 16) |
                  (((unsigned)totrows & 0xFFFFu) << 16));         // dim0 hi | tensor_dim1 lo
    g1[3] = (int)(((unsigned)totrows >> 16) | (64u << 16));       // dim1 hi | tile_dim0=64
    g1[4] = tilerows;                                             // tile_dim1 (tile_dim2=0)
    g1[5] = kdim;                                                 // tensor_dim0_stride[31:0]
    g1[6] = 0;
    g1[7] = 0;
    tdm_g2_t gz = {0, 0, 0, 0};
#if TDM_6ARG
    tdm_g1_t gz8 = {0, 0, 0, 0, 0, 0, 0, 0};
    __builtin_amdgcn_tensor_load_to_lds(g0, g1, gz, gz, gz8, 0);
#else
    __builtin_amdgcn_tensor_load_to_lds(g0, g1, gz, gz, 0);
#endif
}

__device__ __forceinline__ float sigmoidf_(float x) { return 1.f / (1.f + __expf(-x)); }
__device__ __forceinline__ float siluf_(float x)    { return x * sigmoidf_(x); }
__device__ __forceinline__ float softplusf_(float x){ return (x > 20.f) ? x : log1pf(__expf(x)); }

// ---------------- f32 -> bf16 conversion ----------------
__global__ void cvt_bf16_kernel(const float* __restrict__ in, __bf16* __restrict__ out, int n) {
    int i = blockIdx.x * blockDim.x + threadIdx.x;
    if (i < n) out[i] = (__bf16)in[i];
}

// ---------------- bf16 WMMA GEMM with TDM tile feed ----------------
// C[M,N] = A[M,K] * W[N,K]^T ; M%128==0, N%64==0, K%64==0 (true for all calls here)
// Block tile 128x64, K-step 64, double-buffered LDS filled by the Tensor Data Mover.
#define LDS_STRIDE 72   // 64 bf16 + 8 bf16 pad (16B) per row, inserted by TDM pad feature

__global__ __launch_bounds__(256) void gemm_bf16_wmma(
    const __bf16* __restrict__ A, const __bf16* __restrict__ W,
    float* __restrict__ C, int M, int N, int Kdim)
{
    __shared__ __bf16 As[2][128][LDS_STRIDE];   // 36 KB
    __shared__ __bf16 Ws[2][64][LDS_STRIDE];    // 18 KB

    const int tid  = threadIdx.x;
    const int lane = tid & 31;
    const int wave = tid >> 5;              // 0..7
    const int half = lane >> 4;             // 0/1
    const int l16  = lane & 15;
    const int bm = blockIdx.y * 128;
    const int bn = blockIdx.x * 64;
    const int wm = (wave >> 1) * 32;        // 0,32,64,96
    const int wn = (wave & 1) * 32;         // 0,32
    const bool issuer = (tid < 32);         // wave 0 drives the TDM

    v8f acc[2][2] = {};

    if (issuer) {
        tdm_issue_tile(A + (size_t)bm * Kdim, Kdim, M, 128,
                       (unsigned)(size_t)(void*)&As[0][0][0]);
        tdm_issue_tile(W + (size_t)bn * Kdim, Kdim, N, 64,
                       (unsigned)(size_t)(void*)&Ws[0][0][0]);
        __builtin_amdgcn_s_wait_tensorcnt(0);
    }
    __syncthreads();

    int buf = 0;
    for (int k0 = 0; k0 < Kdim; k0 += 64) {
        const int nxt = k0 + 64;
        if (nxt < Kdim && issuer) {
            tdm_issue_tile(A + (size_t)bm * Kdim + nxt, Kdim, M, 128,
                           (unsigned)(size_t)(void*)&As[buf ^ 1][0][0]);
            tdm_issue_tile(W + (size_t)bn * Kdim + nxt, Kdim, N, 64,
                           (unsigned)(size_t)(void*)&Ws[buf ^ 1][0][0]);
        }

        #pragma unroll
        for (int ks = 0; ks < 2; ++ks) {    // two K=32 sub-steps
            Frag16 af[2], bf[2];
            #pragma unroll
            for (int i = 0; i < 2; ++i) {
                // 16-bit A layout: lane row M=l16; K = half*8 + {0..7}, 16+half*8 + {0..7}
                const __bf16* ap = &As[buf][wm + 16 * i + l16][ks * 32 + half * 8];
                af[i].q[0] = *(const v4i*)ap;
                af[i].q[1] = *(const v4i*)(ap + 16);
            }
            #pragma unroll
            for (int j = 0; j < 2; ++j) {
                // 16-bit B layout: lane col N=l16; K = half*8 + {0..7}, 16+half*8 + {0..7}
                const __bf16* bp = &Ws[buf][wn + 16 * j + l16][ks * 32 + half * 8];
                bf[j].q[0] = *(const v4i*)bp;
                bf[j].q[1] = *(const v4i*)(bp + 16);
            }
            #pragma unroll
            for (int i = 0; i < 2; ++i)
                #pragma unroll
                for (int j = 0; j < 2; ++j)
                    acc[i][j] = __builtin_amdgcn_wmma_f32_16x16x32_bf16(
                        false, af[i].v, false, bf[j].v, (short)0, acc[i][j], false, false);
        }

        if (issuer) __builtin_amdgcn_s_wait_tensorcnt(0);
        __syncthreads();
        buf ^= 1;
    }

    // C/D layout: vgpr r -> M = r + half*8, N = l16
    #pragma unroll
    for (int i = 0; i < 2; ++i)
        #pragma unroll
        for (int j = 0; j < 2; ++j) {
            const int col = bn + wn + 16 * j + l16;
            #pragma unroll
            for (int r = 0; r < 8; ++r) {
                const int row = bm + wm + 16 * i + r + half * 8;
                C[(size_t)row * N + col] = acc[i][j][r];
            }
        }
}

// ---------------- causal conv1d (K=4) + SiLU ----------------
__global__ void conv_silu_kernel(const float* __restrict__ proj,
                                 const float* __restrict__ cw,
                                 const float* __restrict__ cb,
                                 float* __restrict__ xBC)
{
    int idx = blockIdx.x * blockDim.x + threadIdx.x;
    if (idx >= Bb * Ll * CONVD) return;
    const int c = idx % CONVD;
    const int t = (idx / CONVD) % Ll;
    const int b = idx / (CONVD * Ll);
    float acc = cb[c];
    #pragma unroll
    for (int k = 0; k < Kk; ++k) {
        const int tt = t + k - (Kk - 1);
        if (tt >= 0)
            acc += proj[(size_t)(b * Ll + tt) * PROJD + Ii + c] * cw[c * Kk + k];
    }
    xBC[(size_t)idx] = siluf_(acc);
}

// ---------------- dt softplus ----------------
__global__ void dt_kernel(const float* __restrict__ proj,
                          const float* __restrict__ dtb,
                          float* __restrict__ dtp)
{
    int idx = blockIdx.x * blockDim.x + threadIdx.x;
    if (idx >= Bb * Ll * Hh) return;
    const int h = idx % Hh;
    dtp[idx] = softplusf_(proj[(size_t)(idx / Hh) * PROJD + (PROJD - Hh) + h] + dtb[h]);
}

// ---------------- conv state output: hbc[:, L-3:, :] -> (B, CONVD, K-1) ----------------
__global__ void conv_state_kernel(const float* __restrict__ proj, float* __restrict__ out)
{
    int idx = blockIdx.x * blockDim.x + threadIdx.x;
    if (idx >= Bb * CONVD * (Kk - 1)) return;
    const int j = idx % (Kk - 1);
    const int c = (idx / (Kk - 1)) % CONVD;
    const int b = idx / ((Kk - 1) * CONVD);
    out[idx] = proj[(size_t)(b * Ll + (Ll - (Kk - 1) + j)) * PROJD + Ii + c];
}

// ---------------- selective scan: one block per (b,h), state in registers ----------------
__global__ __launch_bounds__(256) void scan_kernel(
    const float* __restrict__ xBC, const float* __restrict__ dtp,
    const float* __restrict__ A_log, const float* __restrict__ Dvec,
    float* __restrict__ y, float* __restrict__ fstate)
{
    const int bh = blockIdx.x;
    const int b = bh / Hh, h = bh % Hh;
    const int g = h / (Hh / Gg);
    const int tid = threadIdx.x;
    const int p = tid >> 2;            // 0..63
    const int q = tid & 3;             // 4 lanes share one p-row
    const int nb = q * 32;
    const float Aneg = -__expf(A_log[h]);
    const float Dh = Dvec[h];

    float s[32];
    #pragma unroll
    for (int i = 0; i < 32; ++i) s[i] = 0.f;

    __shared__ float sB[Nn], sC[Nn], sx[Pp];

    for (int t = 0; t < Ll; ++t) {
        const size_t base = (size_t)(b * Ll + t) * CONVD;
        if (tid < Nn) sB[tid]      = xBC[base + Ii + g * Nn + tid];
        else          sC[tid - Nn] = xBC[base + Ii + GNc + g * Nn + (tid - Nn)];
        if (tid < Pp) sx[tid]      = xBC[base + h * Pp + tid];
        __syncthreads();

        const float dt  = dtp[(size_t)(b * Ll + t) * Hh + h];
        const float dA  = __expf(dt * Aneg);
        const float x   = sx[p];
        const float dtx = dt * x;
        float part = 0.f;
        #pragma unroll
        for (int i = 0; i < 32; ++i) {
            const float sv = s[i] * dA + dtx * sB[nb + i];
            s[i] = sv;
            part += sv * sC[nb + i];
        }
        part += __shfl_xor(part, 1, 32);
        part += __shfl_xor(part, 2, 32);
        if (q == 0) y[(size_t)(b * Ll + t) * Ii + h * Pp + p] = part + x * Dh;
        __syncthreads();
    }
    #pragma unroll
    for (int i = 0; i < 32; ++i)
        fstate[((size_t)(b * Hh + h) * Pp + p) * Nn + nb + i] = s[i];
}

// ---------------- grouped RMSNorm + SiLU gate, emit bf16 for out_proj ----------------
__global__ __launch_bounds__(256) void rmsnorm_gate_kernel(
    const float* __restrict__ y, const float* __restrict__ proj,
    const float* __restrict__ nw, __bf16* __restrict__ yn)
{
    const int row  = blockIdx.x;        // b*L + l
    const int tid  = threadIdx.x;
    const int lane = tid & 31;
    const int wave = tid >> 5;          // one wave per group of 512
    const int e0   = wave * GRPSZ + lane * 16;
    const size_t yb = (size_t)row * Ii + e0;

    float v[16];
    float ssq = 0.f;
    #pragma unroll
    for (int i = 0; i < 16; ++i) { v[i] = y[yb + i]; ssq += v[i] * v[i]; }
    #pragma unroll
    for (int m = 1; m < 32; m <<= 1) ssq += __shfl_xor(ssq, m, 32);
    const float rs = rsqrtf(ssq * (1.f / GRPSZ) + EPSI);

    const size_t gb = (size_t)row * PROJD + e0;
    #pragma unroll
    for (int i = 0; i < 16; ++i) {
        const float gate = proj[gb + i];
        yn[yb + i] = (__bf16)(v[i] * rs * nw[e0 + i] * siluf_(gate));
    }
}

// ---------------- host launcher ----------------
extern "C" void kernel_launch(void* const* d_in, const int* in_sizes, int n_in,
                              void* d_out, int out_size, void* d_ws, size_t ws_size,
                              hipStream_t stream)
{
    (void)in_sizes; (void)n_in; (void)out_size; (void)ws_size;
    const float* hs   = (const float*)d_in[0];   // (B,L,D)
    const float* w1   = (const float*)d_in[1];   // (PROJ,D)
    const float* cw   = (const float*)d_in[2];   // (CONVD,K)
    const float* cb   = (const float*)d_in[3];   // (CONVD,)
    const float* dtb  = (const float*)d_in[4];   // (H,)
    const float* alog = (const float*)d_in[5];   // (H,)
    const float* Dv   = (const float*)d_in[6];   // (H,)
    const float* nw   = (const float*)d_in[7];   // (I,)
    const float* w2   = (const float*)d_in[8];   // (D,I)

    float* outp            = (float*)d_out;                          // (B,L,D)
    float* conv_state_out  = outp + (size_t)Bb * Ll * DMODEL;        // (B,CONVD,K-1)
    float* final_state_out = conv_state_out + (size_t)Bb * CONVD * (Kk - 1); // (B,H,P,N)

    char* ws = (char*)d_ws;
    size_t off = 0;
    auto alloc = [&](size_t bytes) -> void* {
        void* p = ws + off;
        off = (off + bytes + 255) & ~(size_t)255;
        return p;
    };
    __bf16* Xbf  = (__bf16*)alloc((size_t)Bb * Ll * DMODEL * 2);   // activations; reused as yn(bf16)
    __bf16* Wbf  = (__bf16*)alloc((size_t)PROJD * DMODEL * 2);     // weights; reused for out_proj
    float*  proj = (float*)alloc((size_t)Bb * Ll * PROJD * 4);
    float*  xBC  = (float*)alloc((size_t)Bb * Ll * CONVD * 4);
    float*  dtpb = (float*)alloc((size_t)Bb * Ll * Hh * 4);
    float*  yb   = (float*)alloc((size_t)Bb * Ll * Ii * 4);

    const int M = Bb * Ll;  // 4096

    // 1) convert activations + in_proj weight to bf16
    int n1 = M * DMODEL;
    cvt_bf16_kernel<<<(n1 + 255) / 256, 256, 0, stream>>>(hs, Xbf, n1);
    int n2 = PROJD * DMODEL;
    cvt_bf16_kernel<<<(n2 + 255) / 256, 256, 0, stream>>>(w1, Wbf, n2);

    // 2) in_proj GEMM: proj = X @ W1^T  (WMMA bf16, f32 acc, TDM tile feed)
    dim3 g1(PROJD / 64, M / 128);
    gemm_bf16_wmma<<<g1, 256, 0, stream>>>(Xbf, Wbf, proj, M, PROJD, DMODEL);

    // 3) conv + silu, dt softplus, conv-state output
    int n3 = Bb * Ll * CONVD;
    conv_silu_kernel<<<(n3 + 255) / 256, 256, 0, stream>>>(proj, cw, cb, xBC);
    int n4 = Bb * Ll * Hh;
    dt_kernel<<<(n4 + 255) / 256, 256, 0, stream>>>(proj, dtb, dtpb);
    int n5 = Bb * CONVD * (Kk - 1);
    conv_state_kernel<<<(n5 + 255) / 256, 256, 0, stream>>>(proj, conv_state_out);

    // 4) selective scan (128 persistent blocks, state in registers)
    scan_kernel<<<Bb * Hh, 256, 0, stream>>>(xBC, dtpb, alog, Dv, yb, final_state_out);

    // 5) grouped RMSNorm + gate -> bf16 (reuse Xbf)
    rmsnorm_gate_kernel<<<M, 256, 0, stream>>>(yb, proj, nw, Xbf);

    // 6) out_proj GEMM: out = yn @ W2^T
    int n6 = DMODEL * Ii;
    cvt_bf16_kernel<<<(n6 + 255) / 256, 256, 0, stream>>>(w2, Wbf, n6);
    dim3 g2(DMODEL / 64, M / 128);
    gemm_bf16_wmma<<<g2, 256, 0, stream>>>(Xbf, Wbf, outp, M, DMODEL, DMODEL);
}
